// ODEfunc_14723147891168
// MI455X (gfx1250) — compile-verified, matrix operand order checked by reading, not run on hardware
//
#include <hip/hip_runtime.h>
#include <hip/hip_bf16.h>

typedef __attribute__((ext_vector_type(16))) _Float16 v16h;
typedef __attribute__((ext_vector_type(8)))  float    v8f;

__device__ __forceinline__ v8f zero8() {
    v8f z = {0.f, 0.f, 0.f, 0.f, 0.f, 0.f, 0.f, 0.f};
    return z;
}

// Hardware tanh (V_TANH_F32 on gfx1250) with safe fallback.
__device__ __forceinline__ float fast_tanh(float x) {
#if __has_builtin(__builtin_amdgcn_tanhf)
    return __builtin_amdgcn_tanhf(x);
#else
    return tanhf(x);
#endif
}

// 16-lane-striped 16-bit fragment load: 8 contiguous halves at base, 8 at base+16.
__device__ __forceinline__ v16h load_frag(const _Float16* p) {
    union { v16h v; uint4 q[2]; } f;
    f.q[0] = *(const uint4*)(p);
    f.q[1] = *(const uint4*)(p + 16);
    return f.v;
}

__device__ __forceinline__ v8f wmma16(v16h a, v16h b, v8f c) {
    return __builtin_amdgcn_wmma_f32_16x16x32_f16(
        false, a, false, b, (short)0, c, false, false);
}

// ---- CDNA5 async global->LDS copy (ASYNCcnt-tracked DMA, no VGPR staging) ----
__device__ __forceinline__ void async_copy16(void* ldsDst, const void* gSrc) {
    unsigned lds = (unsigned)(unsigned long long)ldsDst;   // generic LDS ptr: low 32 = LDS addr
    asm volatile("global_load_async_to_lds_b128 %0, %1, off"
                 :: "v"(lds), "v"(gSrc) : "memory");
}

__device__ __forceinline__ void async_wait0() {
#if __has_builtin(__builtin_amdgcn_s_wait_asynccnt)
    __builtin_amdgcn_s_wait_asynccnt(0);
#else
    asm volatile("s_wait_asynccnt 0x0" ::: "memory");
#endif
}

// Stage a K-slab (NROWS x 32 halves) of a transposed (N-major) f16 weight into LDS
// via async DMA. Padded stride 40 halves so B-fragment loads are contiguous 16B.
template <int NROWS>
__device__ __forceinline__ void stage_slab_async(_Float16* dst,
                                                 const _Float16* __restrict__ srcT,
                                                 int rowLen, int k0, int tid) {
    constexpr int ITER = (NROWS * 4) / 256;   // 4 for 256 rows, 1 for 64 rows
    #pragma unroll
    for (int it = 0; it < ITER; ++it) {
        int idx = it * 256 + tid;
        int n = idx >> 2, q = idx & 3;
        async_copy16(dst + n * 40 + q * 8, srcT + n * rowLen + k0 + q * 8);
    }
}

// ---------------- prep: transposed f16 weights + E = (W3@W1x)^T .* W2 + c1 ----------------
__global__ __launch_bounds__(256) void odefunc_prep(
    const float* __restrict__ t,  const float* __restrict__ W1,
    const float* __restrict__ b1, const float* __restrict__ W2,
    const float* __restrict__ W3,
    _Float16* __restrict__ W1t, _Float16* __restrict__ W2t,
    _Float16* __restrict__ W3t, _Float16* __restrict__ Et,
    float* __restrict__ c1)
{
    const int b = blockIdx.x;    // 0..255
    const int j = threadIdx.x;   // 0..255
    W2t[b * 256 + j] = (_Float16)W2[j * 256 + b];
    float g = 0.f;
    #pragma unroll 8
    for (int d = 0; d < 64; ++d) g += W3[b * 64 + d] * W1[d * 256 + j];
    Et[b * 256 + j] = (_Float16)(W2[j * 256 + b] * g);      // E^T for U = d1 @ E
    if (j < 64) W1t[b * 64 + j] = (_Float16)W1[j * 256 + b];
    if (b < 64) W3t[b * 256 + j] = (_Float16)W3[j * 64 + b];
    if (b == 0) c1[j] = t[0] * W1[64 * 256 + j] + b1[j];
}

// ---------------- main: fused MLP + analytic trace + dv ----------------
#define BM 32

__global__ __launch_bounds__(256) void odefunc_main(
    const float* __restrict__ xaug, const float* __restrict__ b2,
    const float* __restrict__ b3,
    const _Float16* __restrict__ W1t, const _Float16* __restrict__ W2t,
    const _Float16* __restrict__ W3t, const _Float16* __restrict__ Et,
    const float* __restrict__ c1g, float* __restrict__ out)
{
    __shared__ _Float16 xs[32 * 72];           // inp tile (f16)
    __shared__ _Float16 h1s[32 * 264];         // tanh(z1)
    __shared__ _Float16 h2s[32 * 264];         // tanh(z2)
    __shared__ _Float16 wslab[2][256 * 40];    // double-buffered weight K-slab
    __shared__ float trbuf[32];
    __shared__ float dvbuf[32];

    const int tid  = threadIdx.x;
    const int lane = tid & 31;
    const int wave = tid >> 5;
    const int mi   = wave & 1;                 // 16-row tile
    const int ng   = wave >> 1;                // N-group 0..3
    const int ln   = lane & 15;
    const int hi8  = (lane >> 4) << 3;         // 0 or 8
    const int base = blockIdx.x * BM;
    const int arow = mi * 16 + ln;

    const v16h ones16 = { (_Float16)1.f, (_Float16)1.f, (_Float16)1.f, (_Float16)1.f,
                          (_Float16)1.f, (_Float16)1.f, (_Float16)1.f, (_Float16)1.f,
                          (_Float16)1.f, (_Float16)1.f, (_Float16)1.f, (_Float16)1.f,
                          (_Float16)1.f, (_Float16)1.f, (_Float16)1.f, (_Float16)1.f };

    if (tid < 32) { trbuf[tid] = 0.f; dvbuf[tid] = 0.f; }

    // prologue: async-stage GEMM1's first weight slab while we load x
    stage_slab_async<256>(wslab[0], W1t, 64, 0, tid);

    #pragma unroll
    for (int i = 0; i < 8; ++i) {
        int e = i * 256 + tid;
        int r = e >> 6, c = e & 63;
        xs[r * 72 + c] = (_Float16)xaug[(size_t)(base + r) * 66 + c];
    }
    async_wait0();
    __syncthreads();

    // ================= GEMM1: z1 = x @ W1x, K = 64 (2 chunks) =================
    v8f acc1[4] = { zero8(), zero8(), zero8(), zero8() };
    #pragma unroll
    for (int k = 0; k < 2; ++k) {
        if (k + 1 < 2) stage_slab_async<256>(wslab[(k + 1) & 1], W1t, 64, (k + 1) * 32, tid);
        const _Float16* buf = wslab[k & 1];
        v16h a = load_frag(xs + arow * 72 + k * 32 + hi8);
        v16h b0 = load_frag(buf + ((ng * 4 + 0) * 16 + ln) * 40 + hi8);
        v16h b1f = load_frag(buf + ((ng * 4 + 1) * 16 + ln) * 40 + hi8);
        v16h b2f = load_frag(buf + ((ng * 4 + 2) * 16 + ln) * 40 + hi8);
        v16h b3f = load_frag(buf + ((ng * 4 + 3) * 16 + ln) * 40 + hi8);
        acc1[0] = wmma16(a, b0, acc1[0]);
        acc1[1] = wmma16(a, b1f, acc1[1]);
        acc1[2] = wmma16(a, b2f, acc1[2]);
        acc1[3] = wmma16(a, b3f, acc1[3]);
        async_wait0();
        __syncthreads();
    }
    // pre-stage GEMM2 slab 0, then epilogue h1 = tanh(z1 + c1) overlaps the DMA
    stage_slab_async<256>(wslab[0], W2t, 256, 0, tid);
    #pragma unroll
    for (int nt = 0; nt < 4; ++nt) {
        int col = (ng * 4 + nt) * 16 + ln;
        float cv = c1g[col];
        #pragma unroll
        for (int r = 0; r < 8; ++r) {
            int row = mi * 16 + hi8 + r;
            h1s[row * 264 + col] = (_Float16)fast_tanh(acc1[nt][r] + cv);
        }
    }
    async_wait0();
    __syncthreads();

    // ================= GEMM2: z2 = h1 @ W2, K = 256 (8 chunks) =================
    v8f acc2[4] = { zero8(), zero8(), zero8(), zero8() };
    #pragma unroll
    for (int k = 0; k < 8; ++k) {
        if (k + 1 < 8) stage_slab_async<256>(wslab[(k + 1) & 1], W2t, 256, (k + 1) * 32, tid);
        const _Float16* buf = wslab[k & 1];
        v16h a = load_frag(h1s + arow * 264 + k * 32 + hi8);
        v16h b0 = load_frag(buf + ((ng * 4 + 0) * 16 + ln) * 40 + hi8);
        v16h b1f = load_frag(buf + ((ng * 4 + 1) * 16 + ln) * 40 + hi8);
        v16h b2f = load_frag(buf + ((ng * 4 + 2) * 16 + ln) * 40 + hi8);
        v16h b3f = load_frag(buf + ((ng * 4 + 3) * 16 + ln) * 40 + hi8);
        acc2[0] = wmma16(a, b0, acc2[0]);
        acc2[1] = wmma16(a, b1f, acc2[1]);
        acc2[2] = wmma16(a, b2f, acc2[2]);
        acc2[3] = wmma16(a, b3f, acc2[3]);
        async_wait0();
        __syncthreads();
    }
    stage_slab_async<64>(wslab[0], W3t, 256, 0, tid);   // pre-stage GEMM3 slab 0
    #pragma unroll
    for (int nt = 0; nt < 4; ++nt) {
        int col = (ng * 4 + nt) * 16 + ln;
        float bv = b2[col];
        #pragma unroll
        for (int r = 0; r < 8; ++r) {
            int row = mi * 16 + hi8 + r;
            h2s[row * 264 + col] = (_Float16)fast_tanh(acc2[nt][r] + bv);
        }
    }
    async_wait0();
    __syncthreads();

    // ================= GEMM3: dx = h2 @ W3 + b3, N = 64 (8 chunks) =================
    v8f acc3 = zero8();
    #pragma unroll
    for (int k = 0; k < 8; ++k) {
        if (k + 1 < 8) stage_slab_async<64>(wslab[(k + 1) & 1], W3t, 256, (k + 1) * 32, tid);
        const _Float16* buf = wslab[k & 1];
        v16h a = load_frag(h2s + arow * 264 + k * 32 + hi8);
        v16h bfr = load_frag(buf + (ng * 16 + ln) * 40 + hi8);
        acc3 = wmma16(a, bfr, acc3);
        async_wait0();
        __syncthreads();
    }
    stage_slab_async<256>(wslab[0], Et, 256, 0, tid);   // pre-stage GEMM4 slab 0
    {
        int col = ng * 16 + ln;
        float bv = b3[col];
        #pragma unroll
        for (int r = 0; r < 8; ++r) {
            int row = mi * 16 + hi8 + r;
            float dx = acc3[r] + bv;
            out[(size_t)(base + row) * 66 + col] = dx;
            float v = dx * dx;
            v += __shfl_xor(v, 1);
            v += __shfl_xor(v, 2);
            v += __shfl_xor(v, 4);
            v += __shfl_xor(v, 8);
            if (ln == 0) atomicAdd(&dvbuf[row], v);
        }
    }
    async_wait0();
    __syncthreads();

    // ================= GEMM4: U = d1 @ E, tr = rowdot(U, d2), K = 256 =================
    v8f acc4[4] = { zero8(), zero8(), zero8(), zero8() };
    #pragma unroll
    for (int k = 0; k < 8; ++k) {
        if (k + 1 < 8) stage_slab_async<256>(wslab[(k + 1) & 1], Et, 256, (k + 1) * 32, tid);
        const _Float16* buf = wslab[k & 1];
        v16h a = load_frag(h1s + arow * 264 + k * 32 + hi8);
        a = ones16 - a * a;                   // d1 = 1 - h1^2  (packed f16)
        v16h b0 = load_frag(buf + ((ng * 4 + 0) * 16 + ln) * 40 + hi8);
        v16h b1f = load_frag(buf + ((ng * 4 + 1) * 16 + ln) * 40 + hi8);
        v16h b2f = load_frag(buf + ((ng * 4 + 2) * 16 + ln) * 40 + hi8);
        v16h b3f = load_frag(buf + ((ng * 4 + 3) * 16 + ln) * 40 + hi8);
        acc4[0] = wmma16(a, b0, acc4[0]);
        acc4[1] = wmma16(a, b1f, acc4[1]);
        acc4[2] = wmma16(a, b2f, acc4[2]);
        acc4[3] = wmma16(a, b3f, acc4[3]);
        async_wait0();
        __syncthreads();
    }
    {
        float p[8] = {0.f, 0.f, 0.f, 0.f, 0.f, 0.f, 0.f, 0.f};
        #pragma unroll
        for (int nt = 0; nt < 4; ++nt) {
            int col = (ng * 4 + nt) * 16 + ln;
            #pragma unroll
            for (int r = 0; r < 8; ++r) {
                int row = mi * 16 + hi8 + r;
                float hh = (float)h2s[row * 264 + col];
                p[r] += acc4[nt][r] * (1.f - hh * hh);   // d2 = 1 - h2^2
            }
        }
        #pragma unroll
        for (int r = 0; r < 8; ++r) {
            float v = p[r];
            v += __shfl_xor(v, 1);
            v += __shfl_xor(v, 2);
            v += __shfl_xor(v, 4);
            v += __shfl_xor(v, 8);
            if (ln == 0) atomicAdd(&trbuf[mi * 16 + hi8 + r], v);
        }
    }
    __syncthreads();

    if (tid < 32) {
        size_t o = (size_t)(base + tid) * 66;
        out[o + 64] = trbuf[tid];
        out[o + 65] = 0.5f * dvbuf[tid];
    }
}

extern "C" void kernel_launch(void* const* d_in, const int* in_sizes, int n_in,
                              void* d_out, int out_size, void* d_ws, size_t ws_size,
                              hipStream_t stream) {
    const float* t    = (const float*)d_in[0];
    const float* xaug = (const float*)d_in[1];
    const float* W1   = (const float*)d_in[2];
    const float* b1   = (const float*)d_in[3];
    const float* W2   = (const float*)d_in[4];
    const float* b2   = (const float*)d_in[5];
    const float* W3   = (const float*)d_in[6];
    const float* b3   = (const float*)d_in[7];
    float* out = (float*)d_out;

    char* ws = (char*)d_ws;
    _Float16* W1t = (_Float16*)(ws);                               // 32768
    _Float16* W2t = (_Float16*)(ws + 32768);                       // 131072
    _Float16* W3t = (_Float16*)(ws + 32768 + 131072);              // 32768
    _Float16* Et  = (_Float16*)(ws + 32768 + 131072 + 32768);      // 131072
    float*    c1  = (float*)(ws + 32768 + 131072 + 32768 + 131072);

    odefunc_prep<<<256, 256, 0, stream>>>(t, W1, b1, W2, W3, W1t, W2t, W3t, Et, c1);
    odefunc_main<<<16384 / BM, 256, 0, stream>>>(xaug, b2, b3, W1t, W2t, W3t, Et, c1, out);
}
